// QuanvolutionHybrid_65481071395827
// MI455X (gfx1250) — compile-verified, matrix operand order checked by reading, not run on hardware
//
#include <hip/hip_runtime.h>

typedef __attribute__((ext_vector_type(16))) _Float16 v16h;
typedef __attribute__((ext_vector_type(8)))  float    v8f;

#define HW        112
#define NPROW     56
#define NPATCH    3136      // 56*56 patches per image
#define KTOT      50176     // 3136*16
#define BATCH     512
#define NCLS      10
#define WAVES_PB  8
#define CHUNKS    14
#define PP_WAVE   14        // 1568 patch-pairs = 14 chunks * 8 waves * 14

// ---------------------------------------------------------------------------
// Zero the partial-logit accumulator (512 x 16 f32) in workspace.
// ---------------------------------------------------------------------------
__global__ void qh_zero_kernel(float* __restrict__ p, int n) {
    int i = blockIdx.x * blockDim.x + threadIdx.x;
    if (i < n) p[i] = 0.f;
}

// ---------------------------------------------------------------------------
// Fused quanvolution + attention + GEMM partial via v_wmma_f32_16x16x32_f16.
// grid = (32 batch-tiles, 14 k-chunks), block = 256 (8 waves).
// Each wave: 14 k-steps of K=32 (2 patches), accumulates 16x16 f32 tile,
// atomically reduces into acc[512][16].
// ---------------------------------------------------------------------------
__global__ __launch_bounds__(256)
void qh_wmma_kernel(const float* __restrict__ x,
                    const float* __restrict__ Wm,
                    float* __restrict__ acc)
{
    __shared__ float Atile[WAVES_PB][16][32];

    const int lane  = threadIdx.x & 31;
    const int wv    = threadIdx.x >> 5;
    const int mtile = blockIdx.x;                       // 0..31
    const int waveg = blockIdx.y * WAVES_PB + wv;       // 0..111
    const int pp0   = waveg * PP_WAVE;

    const int m  = lane & 15;                           // A row / B column
    const int hi = lane >> 4;
    const int b  = mtile * 16 + m;                      // batch row

    const float co = 0.92387953251128675613f;           // cos(pi/8)
    const float si = 0.38268343236508977173f;           // sin(pi/8)

    v8f accf = {};

    for (int it = 0; it < PP_WAVE; ++it) {
        const int pp = pp0 + it;                        // patch-pair index
        // ---- this lane simulates patch (row b, patch p) ----
        const int p  = pp * 2 + hi;
        const int ph = p / NPROW, pw = p % NPROW;
        const float* xb = x + (size_t)b * (HW * HW) + (2 * ph) * HW + 2 * pw;
        float ang[4] = { xb[0], xb[1], xb[HW], xb[HW + 1] };

        float cc[4], ss[4];
        #pragma unroll
        for (int k = 0; k < 4; ++k) {
            ss[k] = __sinf(ang[k] * 0.5f);
            cc[k] = __cosf(ang[k] * 0.5f);
        }
        float p0[16];
        #pragma unroll
        for (int i = 0; i < 16; ++i) {
            float a = (i & 1 ? ss[0] : cc[0]) * (i & 2 ? ss[1] : cc[1]) *
                      (i & 4 ? ss[2] : cc[2]) * (i & 8 ? ss[3] : cc[3]);
            p0[i] = a * a;
        }
        // CX-chain permutation: final[i] = p0[g01(g12(g23(i)))]
        float fe[16];
        #pragma unroll
        for (int i = 0; i < 16; ++i) {
            int j = i;
            j ^= ((j >> 2) & 1) << 3;
            j ^= ((j >> 1) & 1) << 2;
            j ^= (j & 1) << 1;
            fe[i] = p0[j];
        }
        // ---- attention weight: Ry(fe[0..3]) + CRX(pi/4) chain, P(q0==1) ----
        float c2[4], s2[4];
        #pragma unroll
        for (int k = 0; k < 4; ++k) {
            s2[k] = __sinf(fe[k] * 0.5f);
            c2[k] = __cosf(fe[k] * 0.5f);
        }
        float re[16], im[16];
        #pragma unroll
        for (int i = 0; i < 16; ++i) {
            re[i] = (i & 1 ? s2[0] : c2[0]) * (i & 2 ? s2[1] : c2[1]) *
                    (i & 4 ? s2[2] : c2[2]) * (i & 8 ? s2[3] : c2[3]);
            im[i] = 0.f;
        }
        #pragma unroll
        for (int g = 0; g < 3; ++g) {
            const int cbit = g, tbit = g + 1;
            float nr[16], ni[16];
            #pragma unroll
            for (int i = 0; i < 16; ++i) {
                if ((i >> cbit) & 1) {
                    int pr = i ^ (1 << tbit);
                    nr[i] = co * re[i] + si * im[pr];
                    ni[i] = co * im[i] - si * re[pr];
                } else { nr[i] = re[i]; ni[i] = im[i]; }
            }
            #pragma unroll
            for (int i = 0; i < 16; ++i) { re[i] = nr[i]; im[i] = ni[i]; }
        }
        float w = 0.f;
        #pragma unroll
        for (int i = 1; i < 16; i += 2) w += re[i] * re[i] + im[i] * im[i];

        // ---- stage weighted feats into per-wave LDS A tile [16 m][32 k] ----
        float* dst = &Atile[wv][m][hi * 16];
        #pragma unroll
        for (int j = 0; j < 16; ++j) dst[j] = fe[j] * w;

        // DS ops are wave-wide and in-order: one dscnt wait orders the
        // cross-lane LDS stores before the fragment gather below.
        asm volatile("s_wait_dscnt 0" ::: "memory");

        // ---- A fragment (ISA 16-bit 16x32 layout): K = e + 8*hi (+8 if e>=8)
        const float* rowp = &Atile[wv][m][0];
        v16h av, bv;
        #pragma unroll
        for (int e = 0; e < 8; ++e) av[e]     = (_Float16)rowp[8 * hi + e];
        #pragma unroll
        for (int e = 0; e < 8; ++e) av[8 + e] = (_Float16)rowp[16 + 8 * hi + e];

        // ---- B fragment straight from W[10][50176], N=lane%16, pad N>=10 ----
        const float msk  = (m < NCLS) ? 1.f : 0.f;
        const int   nrow = (m < NCLS) ? m : 0;
        const float* wp  = Wm + (size_t)nrow * KTOT + pp * 32;
        #pragma unroll
        for (int e = 0; e < 8; ++e) bv[e]     = (_Float16)(wp[8 * hi + e] * msk);
        #pragma unroll
        for (int e = 0; e < 8; ++e) bv[8 + e] = (_Float16)(wp[16 + 8 * hi + e] * msk);

        accf = __builtin_amdgcn_wmma_f32_16x16x32_f16(
            false, av, false, bv, (short)0, accf, false, false);
    }

    // C/D layout: VGPR r -> M = r + 8*hi, N = lane%16
    #pragma unroll
    for (int r = 0; r < 8; ++r) {
        int M = r + 8 * hi;
        atomicAdd(&acc[(size_t)(mtile * 16 + M) * 16 + m], accf[r]);
    }
}

// ---------------------------------------------------------------------------
// Bias + log_softmax over 10 classes: out[512][10].
// ---------------------------------------------------------------------------
__global__ void qh_logsoftmax_kernel(const float* __restrict__ acc,
                                     const float* __restrict__ bias,
                                     float* __restrict__ out)
{
    int bi = blockIdx.x * blockDim.x + threadIdx.x;
    if (bi >= BATCH) return;
    float lg[NCLS];
    float mx = -1e30f;
    #pragma unroll
    for (int n = 0; n < NCLS; ++n) {
        lg[n] = acc[bi * 16 + n] + bias[n];
        mx = fmaxf(mx, lg[n]);
    }
    float sum = 0.f;
    #pragma unroll
    for (int n = 0; n < NCLS; ++n) sum += __expf(lg[n] - mx);
    float lse = mx + __logf(sum);
    #pragma unroll
    for (int n = 0; n < NCLS; ++n) out[bi * NCLS + n] = lg[n] - lse;
}

extern "C" void kernel_launch(void* const* d_in, const int* in_sizes, int n_in,
                              void* d_out, int out_size, void* d_ws, size_t ws_size,
                              hipStream_t stream) {
    (void)in_sizes; (void)n_in; (void)out_size; (void)ws_size;
    const float* x    = (const float*)d_in[0];   // [512,1,112,112]
    const float* Wm   = (const float*)d_in[1];   // [10, 50176]
    const float* bias = (const float*)d_in[2];   // [10]
    float* out = (float*)d_out;                  // [512,10]
    float* acc = (float*)d_ws;                   // 512*16 f32 partial logits

    qh_zero_kernel<<<dim3(32), dim3(256), 0, stream>>>(acc, BATCH * 16);
    qh_wmma_kernel<<<dim3(32, CHUNKS), dim3(256), 0, stream>>>(x, Wm, acc);
    qh_logsoftmax_kernel<<<dim3(4), dim3(128), 0, stream>>>(acc, bias, out);
}
